// Sender_84610855731839
// MI455X (gfx1250) — matrix operand order repaired and checked by reading, not compiled, not run on hardware
//
#include <hip/hip_runtime.h>
#include <hip/hip_bf16.h>
#include <math.h>

// ---------------------------------------------------------------------------
// Sender LSTM message generator for MI455X (gfx1250, wave32, WMMA).
// GEMMs in bf16 WMMA (f32 accumulate); double-buffered LDS staging with
// gfx1250 async global->LDS copies when the toolchain exposes them.
// ---------------------------------------------------------------------------

typedef __attribute__((ext_vector_type(16))) __bf16 v16bf;
typedef __attribute__((ext_vector_type(8)))  float  v8f;

union Frag16 { v16bf v; unsigned int u[8]; };

#define B_   1024
#define F_   2048
#define V_   4096
#define E_   256
#define H_   1024
#define H4_  4096
#define KC_  1280   // H + E : catenated [h | emb] K dimension
#define T_   32
#define TP1_ 33
#define TAU_ 1.2f

// gfx1250 async global->LDS copy path (ASYNCcnt-tracked), guarded so the
// file still builds (sync fallback) on toolchains without the builtins.
#if defined(__HIP_DEVICE_COMPILE__) && defined(__has_builtin)
#if __has_builtin(__builtin_amdgcn_global_load_async_to_lds_b128) && \
    __has_builtin(__builtin_amdgcn_s_wait_asynccnt)
#define ASYNC_LDS 1
#endif
#endif
#ifndef ASYNC_LDS
#define ASYNC_LDS 0
#endif

#if ASYNC_LDS
// Builtin signature (from hipcc diagnostic): (v4i AS1*, v4i AS3*, Imm, Imm)
typedef int v4i_ __attribute__((vector_size(16)));
typedef v4i_ __attribute__((address_space(1)))* gv4i_p;
typedef v4i_ __attribute__((address_space(3)))* lv4i_p;
#endif

// ---------------------------------------------------------------------------
// GEMM:  C[m][n] = sum_k A[m][k] * W[n][k] + bias0[n] (+ bias1[n])
// A bf16 [M x lda], W bf16 [N x ldw] (both K-contiguous), C f32 [M x ldc].
// Workgroup tile 128(M) x 256(N); 8 waves (2x4); per-wave 64x64 via 4x4
// v_wmma_f32_16x16x32_bf16. K staged through double-buffered LDS in
// 32-wide slabs (2 x 24 KB of the 320 KB WGP LDS).
// ---------------------------------------------------------------------------
__global__ __launch_bounds__(256)
void gemm_bf16_wmma(const __bf16* __restrict__ A, int lda,
                    const __bf16* __restrict__ W, int ldw,
                    float* __restrict__ C, int ldc, int K,
                    const float* __restrict__ bias0,
                    const float* __restrict__ bias1)
{
    __shared__ __align__(16) __bf16 sA[2][128 * 32];   // 2 x  8 KB
    __shared__ __align__(16) __bf16 sB[2][256 * 32];   // 2 x 16 KB

    const int tid  = threadIdx.x;
    const int lane = tid & 31;
    const int wave = tid >> 5;
    const int wm   = wave >> 2;        // 0..1  (M)
    const int wn   = wave & 3;         // 0..3  (N)
    const int bM   = blockIdx.y * 128;
    const int bN   = blockIdx.x * 256;

    v8f acc[4][4] = {};

    // stage one 128x32 A slab + 256x32 W slab into LDS buffer `buf`
    auto load_tile = [&](int buf, int k0) {
#pragma unroll
        for (int i = 0; i < 2; ++i) {
            int idx = tid + i * 256;
            int r = idx >> 2, cs = idx & 3;
            const __bf16* src = A + (size_t)(bM + r) * lda + k0 + cs * 8;
#if ASYNC_LDS
            __builtin_amdgcn_global_load_async_to_lds_b128(
                (gv4i_p)src, (lv4i_p)&((uint4*)sA[buf])[idx], 0, 0);
#else
            ((uint4*)sA[buf])[idx] = *(const uint4*)src;
#endif
        }
#pragma unroll
        for (int i = 0; i < 4; ++i) {
            int idx = tid + i * 256;
            int r = idx >> 2, cs = idx & 3;
            const __bf16* src = W + (size_t)(bN + r) * ldw + k0 + cs * 8;
#if ASYNC_LDS
            __builtin_amdgcn_global_load_async_to_lds_b128(
                (gv4i_p)src, (lv4i_p)&((uint4*)sB[buf])[idx], 0, 0);
#else
            ((uint4*)sB[buf])[idx] = *(const uint4*)src;
#endif
        }
    };

    const int nk = K >> 5;             // K is a multiple of 32 for all callers
    load_tile(0, 0);

    for (int kk = 0; kk < nk; ++kk) {
        const int cur = kk & 1;
        if (kk + 1 < nk) {
            load_tile(cur ^ 1, (kk + 1) << 5);   // prefetch next slab
#if ASYNC_LDS
            __builtin_amdgcn_s_wait_asynccnt(6); // newest 6 (next slab) in flight
#endif
        } else {
#if ASYNC_LDS
            __builtin_amdgcn_s_wait_asynccnt(0);
#endif
        }
        __syncthreads();                         // current slab visible to all

        const __bf16* pA = sA[cur];
        const __bf16* pB = sB[cur];

        // ---- B fragments: 32(K) x 16(N); lanes 0-15 K=0..15, 16-31 K=16..31
        Frag16 bf[4];
#pragma unroll
        for (int ni = 0; ni < 4; ++ni) {
            int n = wn * 64 + ni * 16 + (lane & 15);
            const __bf16* p = &pB[n * 32 + (lane >> 4) * 16];
#pragma unroll
            for (int j = 0; j < 8; ++j)
                bf[ni].u[j] = *(const unsigned int*)(p + 2 * j);
        }
        // ---- A fragments: 16(M) x 32(K); lanes 0-15 K=0..15, 16-31 K=16..31
#pragma unroll
        for (int mi = 0; mi < 4; ++mi) {
            Frag16 af;
            int m = wm * 64 + mi * 16 + (lane & 15);
            const __bf16* p = &pA[m * 32 + (lane >> 4) * 16];
#pragma unroll
            for (int j = 0; j < 4; ++j)
                af.u[j] = *(const unsigned int*)(p + 2 * j);
#pragma unroll
            for (int j = 0; j < 4; ++j)
                af.u[4 + j] = *(const unsigned int*)(p + 8 + 2 * j);
#pragma unroll
            for (int ni = 0; ni < 4; ++ni) {
                acc[mi][ni] = __builtin_amdgcn_wmma_f32_16x16x32_bf16(
                    false, af.v, false, bf[ni].v,
                    (short)0, acc[mi][ni], false, false);
            }
        }
        __syncthreads();                         // done reading `cur` buffer
    }

    // ---- epilogue: C/D layout VGPR r -> M = r + 8*(lane>=16), N = lane%16
#pragma unroll
    for (int ni = 0; ni < 4; ++ni) {
        int n = bN + wn * 64 + ni * 16 + (lane & 15);
        float bv = bias0 ? bias0[n] : 0.0f;
        if (bias1) bv += bias1[n];
#pragma unroll
        for (int mi = 0; mi < 4; ++mi) {
            int mrow = bM + wm * 64 + mi * 16 + ((lane >> 4) << 3);
            v8f a = acc[mi][ni];
#pragma unroll
            for (int r = 0; r < 8; ++r)
                C[(size_t)(mrow + r) * ldc + n] = a[r] + bv;
        }
    }
}

// ---------------------------------------------------------------------------
// Elementwise / helper kernels
// ---------------------------------------------------------------------------
__global__ __launch_bounds__(256)
void cvt_bf16(const float* __restrict__ src, __bf16* __restrict__ dst, int n)
{
    int i = blockIdx.x * 256 + threadIdx.x;
    if (i < n) dst[i] = (__bf16)src[i];
}

// Wcat[n][k] = k<H ? w_hh[n][k] : w_ih[n][k-H]   (bf16, row stride KC_)
__global__ __launch_bounds__(256)
void build_wcat(const float* __restrict__ w_hh, const float* __restrict__ w_ih,
                __bf16* __restrict__ Wc)
{
    int i = blockIdx.x * 256 + threadIdx.x;
    if (i >= H4_ * KC_) return;
    int n = i / KC_, k = i - n * KC_;
    float v = (k < H_) ? w_hh[(size_t)n * H_ + k]
                       : w_ih[(size_t)n * E_ + (k - H_)];
    Wc[i] = (__bf16)v;
}

__device__ __forceinline__ float sigm(float x) { return 1.0f / (1.0f + expf(-x)); }

// c = sig(f)*c + sig(i)*tanh(g); h = sig(o)*tanh(c); X[:, :H] = bf16(h)
__global__ __launch_bounds__(256)
void lstm_cell(const float* __restrict__ gates,
               float* __restrict__ h, float* __restrict__ c,
               __bf16* __restrict__ X)
{
    const int b = blockIdx.x, tid = threadIdx.x;
    const float* g = gates + (size_t)b * H4_;
#pragma unroll
    for (int it = 0; it < H_ / 256; ++it) {
        int j = tid + it * 256;
        float gi = g[j], gf = g[H_ + j], gg = g[2 * H_ + j], go = g[3 * H_ + j];
        float cc = sigm(gf) * c[(size_t)b * H_ + j] + sigm(gi) * tanhf(gg);
        float hh = sigm(go) * tanhf(cc);
        c[(size_t)b * H_ + j] = cc;
        h[(size_t)b * H_ + j] = hh;
        X[(size_t)b * KC_ + j] = (__bf16)hh;
    }
}

// After h0 GEMM: zero c, X[:, :H]=bf16(h0), X[:, H:]=bf16(emb[start]),
// out[:,0,:] = one_hot(start)   (output is streamed non-temporally)
__global__ __launch_bounds__(256)
void init_state(const float* __restrict__ h0, float* __restrict__ c,
                __bf16* __restrict__ X, const float* __restrict__ embedding,
                const int* __restrict__ start_idx, float* __restrict__ out)
{
    const int b = blockIdx.x, tid = threadIdx.x;
    const int sidx = *start_idx;
#pragma unroll
    for (int i = 0; i < H_ / 256; ++i) {
        int j = tid + i * 256;
        c[(size_t)b * H_ + j] = 0.0f;
        X[(size_t)b * KC_ + j] = (__bf16)h0[(size_t)b * H_ + j];
    }
    X[(size_t)b * KC_ + H_ + tid] = (__bf16)embedding[(size_t)sidx * E_ + tid];
#pragma unroll
    for (int i = 0; i < V_ / 256; ++i) {
        int v = tid + i * 256;
        __builtin_nontemporal_store((v == sidx) ? 1.0f : 0.0f,
                                    &out[(size_t)b * TP1_ * V_ + v]);
    }
}

// ---------------------------------------------------------------------------
// Fused log_softmax -> gumbel -> softmax -> argmax -> straight-through token
// -> scaled embedding row gather into X[:, H:].  One block per batch row.
// ---------------------------------------------------------------------------
__device__ __forceinline__ float gumbel_noise(int step, int b, int v)
{
    unsigned s = (unsigned)(step + 1) * 0x9E3779B9u
               ^ (unsigned)b * 0x85EBCA6Bu
               ^ (unsigned)v * 0xC2B2AE35u;
    s ^= s >> 16; s *= 0x7FEB352Du;
    s ^= s >> 15; s *= 0x846CA68Bu;
    s ^= s >> 16;
    float u = (float)(s >> 8) * (1.0f / 16777216.0f) + 5.96e-8f;
    return -logf(-logf(u));
}

__global__ __launch_bounds__(256)
void softmax_sample(const float* __restrict__ logits,
                    const float* __restrict__ embedding,
                    float* __restrict__ out,
                    __bf16* __restrict__ X, int step)
{
    __shared__ float zrow[V_];       // 16 KB of the 320 KB WGP LDS
    __shared__ float redf[8];
    __shared__ int   redi[8];
    __shared__ float s_scale;

    const int b = blockIdx.x, tid = threadIdx.x;
    const int lane = tid & 31, wave = tid >> 5;
    const float* lrow = logits + (size_t)b * V_;

    // 1) row max
    float mx = -3.4e38f;
#pragma unroll
    for (int i = 0; i < V_ / 256; ++i) mx = fmaxf(mx, lrow[tid + i * 256]);
#pragma unroll
    for (int off = 16; off > 0; off >>= 1) mx = fmaxf(mx, __shfl_xor(mx, off, 32));
    if (lane == 0) redf[wave] = mx;
    __syncthreads();
    if (tid == 0) {
        float m = redf[0];
        for (int i = 1; i < 8; ++i) m = fmaxf(m, redf[i]);
        redf[0] = m;
    }
    __syncthreads();
    const float rowmax = redf[0];
    __syncthreads();

    // 2) log-sum-exp
    float sm = 0.0f;
#pragma unroll
    for (int i = 0; i < V_ / 256; ++i) sm += expf(lrow[tid + i * 256] - rowmax);
#pragma unroll
    for (int off = 16; off > 0; off >>= 1) sm += __shfl_xor(sm, off, 32);
    if (lane == 0) redf[wave] = sm;
    __syncthreads();
    if (tid == 0) {
        float s = 0.0f;
        for (int i = 0; i < 8; ++i) s += redf[i];
        redf[0] = s;
    }
    __syncthreads();
    const float lse = rowmax + logf(redf[0]);
    __syncthreads();

    // 3) z = (logp + gumbel)/tau, track argmax
    float zm = -3.4e38f; int zi = 0;
#pragma unroll
    for (int i = 0; i < V_ / 256; ++i) {
        int v = tid + i * 256;
        float z = (lrow[v] - lse + gumbel_noise(step, b, v)) * (1.0f / TAU_);
        zrow[v] = z;
        if (z > zm) { zm = z; zi = v; }
    }
#pragma unroll
    for (int off = 16; off > 0; off >>= 1) {
        float oz = __shfl_xor(zm, off, 32);
        int   oi = __shfl_xor(zi, off, 32);
        if (oz > zm || (oz == zm && oi < zi)) { zm = oz; zi = oi; }
    }
    if (lane == 0) { redf[wave] = zm; redi[wave] = zi; }
    __syncthreads();
    if (tid == 0) {
        float m = redf[0]; int ix = redi[0];
        for (int i = 1; i < 8; ++i)
            if (redf[i] > m || (redf[i] == m && redi[i] < ix)) { m = redf[i]; ix = redi[i]; }
        redf[0] = m; redi[0] = ix;
    }
    __syncthreads();
    const float zmax = redf[0];
    const int   amax = redi[0];
    __syncthreads();

    // 4) softmax denominator over z
    float sz = 0.0f;
#pragma unroll
    for (int i = 0; i < V_ / 256; ++i) sz += expf(zrow[tid + i * 256] - zmax);
#pragma unroll
    for (int off = 16; off > 0; off >>= 1) sz += __shfl_xor(sz, off, 32);
    if (lane == 0) redf[wave] = sz;
    __syncthreads();
    if (tid == 0) {
        float s = 0.0f;
        for (int i = 0; i < 8; ++i) s += redf[i];
        redf[0] = s;
    }
    __syncthreads();
    const float inv = 1.0f / redf[0];

    // 5) straight-through token slab: exactly 0 off-argmax.
    //    Output is written once and never read on-device -> non-temporal.
    float* orow = out + (size_t)b * TP1_ * V_ + (size_t)(step + 1) * V_;
#pragma unroll
    for (int i = 0; i < V_ / 256; ++i) {
        int v = tid + i * 256;
        float y = expf(zrow[v] - zmax) * inv;
        float hard = (v == amax) ? 1.0f : 0.0f;
        float tok = (hard - y) + y;
        __builtin_nontemporal_store(tok, &orow[v]);
        if (v == amax) s_scale = tok;
    }
    __syncthreads();

    // 6) emb = tok[amax] * embedding[amax]  (exact: all other entries are 0)
    X[(size_t)b * KC_ + H_ + tid] =
        (__bf16)(s_scale * embedding[(size_t)amax * E_ + tid]);
}

// ---------------------------------------------------------------------------
// Host launch sequence
// ---------------------------------------------------------------------------
extern "C" void kernel_launch(void* const* d_in, const int* in_sizes, int n_in,
                              void* d_out, int out_size, void* d_ws, size_t ws_size,
                              hipStream_t stream)
{
    (void)in_sizes; (void)n_in; (void)out_size; (void)ws_size;

    const float* t     = (const float*)d_in[0];
    const float* emb   = (const float*)d_in[1];
    const float* aff_w = (const float*)d_in[2];
    const float* aff_b = (const float*)d_in[3];
    const float* w_ih  = (const float*)d_in[4];
    const float* w_hh  = (const float*)d_in[5];
    const float* b_ih  = (const float*)d_in[6];
    const float* b_hh  = (const float*)d_in[7];
    const float* lp_w  = (const float*)d_in[8];
    const float* lp_b  = (const float*)d_in[9];
    const int*   sidx  = (const int*)d_in[10];
    float* out = (float*)d_out;

    char* p = (char*)d_ws;
    auto take = [&](size_t bytes) {
        char* q = p;
        p += (bytes + 255) & ~(size_t)255;
        return q;
    };
    __bf16* Xbf  = (__bf16*)take((size_t)B_ * KC_ * 2);   // [B, H+E]  activations
    __bf16* Wcat = (__bf16*)take((size_t)H4_ * KC_ * 2);  // [4H, H+E] cat weights
    __bf16* LPbf = (__bf16*)take((size_t)V_ * H_ * 2);    // [V, H]
    __bf16* Tbf  = (__bf16*)take((size_t)B_ * F_ * 2);    // [B, F]
    __bf16* AWbf = (__bf16*)take((size_t)H_ * F_ * 2);    // [H, F]
    float*  gate = (float*)take((size_t)B_ * H4_ * 4);    // [B, 4H]
    float*  logi = (float*)take((size_t)B_ * V_ * 4);     // [B, V]
    float*  hbuf = (float*)take((size_t)B_ * H_ * 4);
    float*  cbuf = (float*)take((size_t)B_ * H_ * 4);

    // one-time (per call) weight/activation precision conversion
    cvt_bf16<<<(B_ * F_) / 256, 256, 0, stream>>>(t, Tbf, B_ * F_);
    cvt_bf16<<<(H_ * F_) / 256, 256, 0, stream>>>(aff_w, AWbf, H_ * F_);
    cvt_bf16<<<(V_ * H_) / 256, 256, 0, stream>>>(lp_w, LPbf, V_ * H_);
    build_wcat<<<(H4_ * KC_) / 256, 256, 0, stream>>>(w_hh, w_ih, Wcat);

    // h0 = t @ aff_w.T + aff_b
    gemm_bf16_wmma<<<dim3(H_ / 256, B_ / 128), 256, 0, stream>>>(
        Tbf, F_, AWbf, F_, hbuf, H_, F_, aff_b, nullptr);
    init_state<<<B_, 256, 0, stream>>>(hbuf, cbuf, Xbf, emb, sidx, out);

    for (int step = 0; step < T_; ++step) {
        // gates = [h|emb] @ [w_hh|w_ih].T + b_ih + b_hh
        gemm_bf16_wmma<<<dim3(H4_ / 256, B_ / 128), 256, 0, stream>>>(
            Xbf, KC_, Wcat, KC_, gate, H4_, KC_, b_ih, b_hh);
        lstm_cell<<<B_, 256, 0, stream>>>(gate, hbuf, cbuf, Xbf);
        // logits = h @ lp_w.T + lp_b   (A = X[:, :H], lda = KC_)
        gemm_bf16_wmma<<<dim3(V_ / 256, B_ / 128), 256, 0, stream>>>(
            Xbf, KC_, LPbf, H_, logi, V_, H_, lp_b, nullptr);
        softmax_sample<<<B_, 256, 0, stream>>>(logi, emb, out, Xbf, step);
    }
}
